// TransformerDecoderLayer_7653631721862
// MI455X (gfx1250) — compile-verified, matrix operand order checked by reading, not run on hardware
//
#include <hip/hip_runtime.h>
#include <hip/hip_bf16.h>
#include <cstdint>

// ---------------------------------------------------------------------------
// Deformable-DETR decoder layer for MI455X (gfx1250, wave32, WMMA + TDM).
//
// Roofline: ~9 GFLOP of GEMMs (trivial for v_wmma_f32_16x16x32_f16) vs
// ~125 MB of value-tensor gather traffic -> memory bound (~6us floor at
// 23.3 TB/s, less with 192MB L2). Design choices:
//   * all GEMM operands pre-packed to f16 once (halves re-read traffic),
//   * GEMM tiles moved global->LDS by the Tensor Data Mover (TENSORcnt,
//     double-buffered, pad feature gives the 40-half padded LDS stride),
//   * gather kernel: lane == head-dim channel (HEAD_DIM==32==wave32) for
//     fully coalesced 128B corner fetches, wave-uniform control flow.
// ---------------------------------------------------------------------------

typedef __attribute__((ext_vector_type(16))) _Float16 v16h;
typedef __attribute__((ext_vector_type(4)))  _Float16 v4h;
typedef __attribute__((ext_vector_type(8)))  float    v8f;
typedef __attribute__((ext_vector_type(4)))  unsigned int u32x4;
typedef __attribute__((ext_vector_type(8)))  int      i32x8;
typedef __attribute__((ext_vector_type(4)))  int      i32x4;

static constexpr int BQ    = 16;      // batch
static constexpr int QL    = 300;     // queries
static constexpr int DM    = 256;     // d_model
static constexpr int NH    = 8;       // heads
static constexpr int LENV  = 8500;    // sum of level sizes
static constexpr int MROWS = BQ * QL; // 4800

static __device__ __forceinline__ int imin(int a, int b) { return a < b ? a : b; }

#if defined(__has_builtin)
#if __has_builtin(__builtin_amdgcn_tensor_load_to_lds) && \
    __has_builtin(__builtin_amdgcn_s_wait_tensorcnt)
#define USE_TDM 1
#endif
#endif

#ifdef USE_TDM
// Issue one TDM 2D tile load: tile_dim0=32 elems (f16), tile_dim1=64 rows,
// row stride K elems, LDS padded +4 DWORDs every 16 DWORDs (stride 40 halves).
static __device__ __forceinline__ void
tdm_load_tile(uint32_t lds_addr, const _Float16* gptr, int K, int nrows)
{
    u32x4 g0;
    g0[0] = 1u;                                   // count=1, user mode
    g0[1] = lds_addr;                             // lds_addr [63:32]
    const uint64_t ga = (uint64_t)(uintptr_t)gptr;
    g0[2] = (uint32_t)ga;                         // global_addr [95:64]
    g0[3] = ((uint32_t)(ga >> 32) & 0x1FFFFFFu)   // global_addr [120:96]
            | (2u << 30);                         // type=2 (image) [127:126]

    i32x8 g1;
    g1[0] = (int)((1u << 16)                      // data_size=1 (2 bytes)
                  | (1u << 20)                    // pad_enable
                  | (3u << 22)                    // pad_interval: 16 DWORDs
                  | (3u << 25));                  // pad_amount: 4 DWORDs
    g1[1] = (int)(((uint32_t)K & 0xFFFFu) << 16);            // tensor_dim0 lo
    g1[2] = (int)((((uint32_t)K >> 16) & 0xFFFFu)            // tensor_dim0 hi
                  | (((uint32_t)nrows & 0xFFFFu) << 16));    // tensor_dim1 lo
    g1[3] = (int)(((((uint32_t)nrows >> 16) & 0xFFFFu))      // tensor_dim1 hi
                  | (32u << 16));                            // tile_dim0 = 32
    g1[4] = (int)(64u);                                      // tile_dim1=64
    g1[5] = K;                                               // dim0_stride lo
    g1[6] = 0;
    g1[7] = 0;
    const i32x4 z4 = {0, 0, 0, 0};
    const i32x8 z8 = {0, 0, 0, 0, 0, 0, 0, 0};
    // 6-arg toolchain signature: (g0, g1, g2, g3, g_extra, cpol)
    __builtin_amdgcn_tensor_load_to_lds(g0, g1, z4, z4, z8, 0);
}
#endif

// ---------------------------------------------------------------------------
// Pack f32 (optionally sum of two f32 tensors) into f16.
// ---------------------------------------------------------------------------
__global__ __launch_bounds__(256)
void pack_f16_kernel(const float* __restrict__ a, const float* __restrict__ b,
                     _Float16* __restrict__ out, int total)
{
    const int i = (blockIdx.x * 256 + threadIdx.x) * 4;
    if (i < total) {
        float4 v = *(const float4*)(a + i);
        if (b) {
            float4 p = *(const float4*)(b + i);
            v.x += p.x; v.y += p.y; v.z += p.z; v.w += p.w;
        }
        v4h h;
        h[0] = (_Float16)v.x; h[1] = (_Float16)v.y;
        h[2] = (_Float16)v.z; h[3] = (_Float16)v.w;
        *(v4h*)(out + i) = h;
    }
}

// ---------------------------------------------------------------------------
// Tiled GEMM: C[M,N] = Ah[M,K] * Wh[N,K]^T + bias[N]  (f16 in, f32 out)
// Block tile 64x64, 256 threads = 8 waves (4x2), wave tile 16x32.
// Tiles land in LDS via TDM (double-buffered) or manual staging fallback.
// ---------------------------------------------------------------------------
__global__ __launch_bounds__(256)
void gemm_bias_wmma(const _Float16* __restrict__ Ah, const _Float16* __restrict__ Wh,
                    const float* __restrict__ bias, float* __restrict__ C,
                    int N, int K, int Mtot)
{
    __shared__ _Float16 As[2][64][40];
    __shared__ _Float16 Bs[2][64][40];

    const int t    = threadIdx.x;
    const int wave = t >> 5;
    const int lane = t & 31;
    const int wm   = wave >> 1;
    const int wn   = wave & 1;
    const int m0   = blockIdx.y * 64;
    const int n0   = blockIdx.x * 64;
    const int fr   = lane & 15;
    const int kb   = (lane & 16) ? 8 : 0;
    const int S    = K >> 5;

    v8f c0 = {}; v8f c1 = {};

#ifdef USE_TDM
    const uint32_t ldsA = (uint32_t)(uintptr_t)&As[0][0][0];
    const uint32_t ldsB = (uint32_t)(uintptr_t)&Bs[0][0][0];
    const uint32_t bufBytes = 64 * 40 * 2;
    if (wave == 0) {     // one DMA issuer per block; TENSORcnt is per-wave
        tdm_load_tile(ldsA, Ah + (size_t)m0 * K, K, Mtot);
        tdm_load_tile(ldsB, Wh + (size_t)n0 * K, K, N);
    }
#else
    const int lr = t >> 2;
    const int lc = (t & 3) * 8;
#endif

    for (int s = 0; s < S; s++) {
        const int cur = s & 1;
#ifdef USE_TDM
        if (wave == 0) {
            if (s + 1 < S) {     // prefetch next tile, then wait current
                const int nxt = (s + 1) & 1;
                tdm_load_tile(ldsA + nxt * bufBytes,
                              Ah + (size_t)m0 * K + (s + 1) * 32, K, Mtot);
                tdm_load_tile(ldsB + nxt * bufBytes,
                              Wh + (size_t)n0 * K + (s + 1) * 32, K, N);
                __builtin_amdgcn_s_wait_tensorcnt((short)2);
            } else {
                __builtin_amdgcn_s_wait_tensorcnt((short)0);
            }
        }
#else
        {   // manual staging fallback: 8 halves per thread per tile
            const _Float16* ap = Ah + (size_t)(m0 + lr) * K + s * 32 + lc;
            const _Float16* wp = Wh + (size_t)(n0 + lr) * K + s * 32 + lc;
            *(v4h*)&As[cur][lr][lc]     = *(const v4h*)ap;
            *(v4h*)&As[cur][lr][lc + 4] = *(const v4h*)(ap + 4);
            *(v4h*)&Bs[cur][lr][lc]     = *(const v4h*)wp;
            *(v4h*)&Bs[cur][lr][lc + 4] = *(const v4h*)(wp + 4);
            if (s + 1 < S) {
                __builtin_prefetch(ap + 32, 0, 1);
                __builtin_prefetch(wp + 32, 0, 1);
            }
        }
#endif
        __syncthreads();

        v16h af, b0f, b1f;
        #pragma unroll
        for (int i = 0; i < 8; i++) {
            af[i]      = As[cur][wm * 16 + fr][kb + i];
            af[8 + i]  = As[cur][wm * 16 + fr][kb + 16 + i];
            b0f[i]     = Bs[cur][wn * 32 + fr][kb + i];
            b0f[8 + i] = Bs[cur][wn * 32 + fr][kb + 16 + i];
            b1f[i]     = Bs[cur][wn * 32 + 16 + fr][kb + i];
            b1f[8 + i] = Bs[cur][wn * 32 + 16 + fr][kb + 16 + i];
        }
        c0 = __builtin_amdgcn_wmma_f32_16x16x32_f16(false, af, false, b0f,
                                                    (short)0, c0, false, false);
        c1 = __builtin_amdgcn_wmma_f32_16x16x32_f16(false, af, false, b1f,
                                                    (short)0, c1, false, false);
        __syncthreads();
    }

    const int col0  = n0 + wn * 32 + fr;
    const float bi0 = bias[col0];
    const float bi1 = bias[col0 + 16];
    const int radd  = (lane & 16) ? 8 : 0;
    #pragma unroll
    for (int r = 0; r < 8; r++) {
        const size_t row = (size_t)(m0 + wm * 16 + r + radd);
        C[row * N + col0]      = c0[r] + bi0;
        C[row * N + col0 + 16] = c1[r] + bi1;
    }
}

// ---------------------------------------------------------------------------
// Multi-head self-attention: one (b, h, 16-query tile) per 64-thread block.
// scores = QK^T/sqrt(32) via WMMA, f32 softmax in LDS, P*V via WMMA.
// Writes f16 output directly (feeds the out-projection GEMM).
// ---------------------------------------------------------------------------
__global__ __launch_bounds__(64)
void mha_kernel(const float* __restrict__ Qb, const float* __restrict__ Kb,
                const float* __restrict__ Vb, _Float16* __restrict__ SA)
{
    __shared__ _Float16 Qs[16][40];
    __shared__ _Float16 Ks[2][16][40];
    __shared__ float    S[16][312];
    __shared__ _Float16 P[16][328];

    const int b  = blockIdx.z;
    const int h  = blockIdx.y;
    const int q0 = blockIdx.x * 16;
    const int t  = threadIdx.x;
    const int w  = t >> 5;
    const int lane = t & 31;
    const int fr   = lane & 15;
    const int kb   = (lane & 16) ? 8 : 0;
    const int radd = (lane & 16) ? 8 : 0;

    {
        const int row = t >> 2, col = (t & 3) * 8;
        const int gq = imin(q0 + row, QL - 1);
        const float* qp = Qb + (size_t)(b * QL + gq) * DM + h * 32 + col;
        float4 v0 = *(const float4*)qp;
        float4 v1 = *(const float4*)(qp + 4);
        Qs[row][col + 0] = (_Float16)v0.x; Qs[row][col + 1] = (_Float16)v0.y;
        Qs[row][col + 2] = (_Float16)v0.z; Qs[row][col + 3] = (_Float16)v0.w;
        Qs[row][col + 4] = (_Float16)v1.x; Qs[row][col + 5] = (_Float16)v1.y;
        Qs[row][col + 6] = (_Float16)v1.z; Qs[row][col + 7] = (_Float16)v1.w;
    }
    __syncthreads();

    v16h qf;
    #pragma unroll
    for (int i = 0; i < 8; i++) {
        qf[i]     = Qs[fr][kb + i];
        qf[8 + i] = Qs[fr][kb + 16 + i];
    }

    const float scale = 0.17677669529663687f;  // 1/sqrt(32)

    for (int jj = 0; jj < 10; jj++) {
        const int j = jj * 2 + w;
        const bool act = (j < 19);
        if (act) {
            const int krow = lane >> 1, kcol = (lane & 1) * 16;
            const int gk = imin(j * 16 + krow, QL - 1);
            const float* kp = Kb + (size_t)(b * QL + gk) * DM + h * 32 + kcol;
            #pragma unroll
            for (int c = 0; c < 16; c += 4) {
                float4 v = *(const float4*)(kp + c);
                Ks[w][krow][kcol + c + 0] = (_Float16)v.x;
                Ks[w][krow][kcol + c + 1] = (_Float16)v.y;
                Ks[w][krow][kcol + c + 2] = (_Float16)v.z;
                Ks[w][krow][kcol + c + 3] = (_Float16)v.w;
            }
        }
        __syncthreads();   // uniform trip count: both waves always arrive
        if (act) {
            v16h kf;
            #pragma unroll
            for (int i = 0; i < 8; i++) {
                kf[i]     = Ks[w][fr][kb + i];
                kf[8 + i] = Ks[w][fr][kb + 16 + i];
            }
            v8f sc = {};
            sc = __builtin_amdgcn_wmma_f32_16x16x32_f16(false, qf, false, kf,
                                                        (short)0, sc, false, false);
            const int col = j * 16 + fr;
            #pragma unroll
            for (int r = 0; r < 8; r++)
                S[r + radd][col] = (col < QL) ? sc[r] * scale : -1e30f;
        }
    }
    __syncthreads();

    {
        const int row = t >> 2, sub = t & 3;
        float mx = -3e38f;
        for (int c = sub; c < 304; c += 4) mx = fmaxf(mx, S[row][c]);
        mx = fmaxf(mx, __shfl_xor(mx, 1, 32));
        mx = fmaxf(mx, __shfl_xor(mx, 2, 32));
        float sum = 0.f;
        for (int c = sub; c < 304; c += 4) {
            const float e = __expf(S[row][c] - mx);
            S[row][c] = e;
            sum += e;
        }
        sum += __shfl_xor(sum, 1, 32);
        sum += __shfl_xor(sum, 2, 32);
        const float inv = 1.0f / sum;
        for (int c = sub; c < 304; c += 4) P[row][c] = (_Float16)(S[row][c] * inv);
        for (int c = 304 + sub; c < 320; c += 4) P[row][c] = (_Float16)0.f;
    }
    __syncthreads();

    v8f acc = {};
    for (int kt = 0; kt < 10; kt++) {
        v16h pf, vf;
        #pragma unroll
        for (int i = 0; i < 8; i++) {
            pf[i]     = P[fr][kt * 32 + kb + i];
            pf[8 + i] = P[fr][kt * 32 + kb + 16 + i];
        }
        #pragma unroll
        for (int i = 0; i < 8; i++) {
            const int kr0 = imin(kt * 32 + kb + i, QL - 1);
            const int kr1 = imin(kt * 32 + kb + 16 + i, QL - 1);
            vf[i]     = (_Float16)Vb[(size_t)(b * QL + kr0) * DM + h * 32 + w * 16 + fr];
            vf[8 + i] = (_Float16)Vb[(size_t)(b * QL + kr1) * DM + h * 32 + w * 16 + fr];
        }
        acc = __builtin_amdgcn_wmma_f32_16x16x32_f16(false, pf, false, vf,
                                                     (short)0, acc, false, false);
    }
    #pragma unroll
    for (int r = 0; r < 8; r++) {
        const int q = q0 + r + radd;
        if (q < QL)
            SA[(size_t)(b * QL + q) * DM + h * 32 + w * 16 + fr] = (_Float16)acc[r];
    }
}

// ---------------------------------------------------------------------------
// Deformable attention gather: one wave per (b,q,h); lane == channel (32).
// ---------------------------------------------------------------------------
__global__ __launch_bounds__(256)
void deform_attn(const float* __restrict__ offb, const float* __restrict__ awb,
                 const float* __restrict__ ref, const float* __restrict__ value,
                 float* __restrict__ ca)
{
    const int wid  = blockIdx.x * 8 + (threadIdx.x >> 5);
    const int lane = threadIdx.x & 31;
    const int b    = wid / (QL * NH);
    const int rem  = wid % (QL * NH);
    const int q    = rem / NH;
    const int h    = rem % NH;
    const size_t row = (size_t)(b * QL + q);

    const float offv = offb[row * DM + h * 32 + lane];

    float awv = (lane < 16) ? awb[row * 128 + h * 16 + lane] : -3e38f;
    float m = awv;
    m = fmaxf(m, __shfl_xor(m, 8, 16));
    m = fmaxf(m, __shfl_xor(m, 4, 16));
    m = fmaxf(m, __shfl_xor(m, 2, 16));
    m = fmaxf(m, __shfl_xor(m, 1, 16));
    float e = __expf(awv - m);
    float s = e;
    s += __shfl_xor(s, 8, 16);
    s += __shfl_xor(s, 4, 16);
    s += __shfl_xor(s, 2, 16);
    s += __shfl_xor(s, 1, 16);
    const float wnorm = e / s;

    const float r0 = ref[row * 4 + 0], r1 = ref[row * 4 + 1];
    const float r2 = ref[row * 4 + 2], r3 = ref[row * 4 + 3];

    const int lvlH[4] = {80, 40, 20, 10};
    const int lvlW[4] = {80, 40, 20, 10};
    const int lvlS[4] = {0, 6400, 8000, 8400};

    float acc = 0.f;
    #pragma unroll
    for (int p = 0; p < 16; p++) {
        const int lvl = p >> 2;
        const int Hh = lvlH[lvl], Ww = lvlW[lvl], st = lvlS[lvl];
        const float ox = __shfl(offv, 2 * p, 32);
        const float oy = __shfl(offv, 2 * p + 1, 32);
        const float wp = __shfl(wnorm, p, 32);
        const float x = (r0 + ox * 0.125f * r2) * (float)Ww - 0.5f;
        const float y = (r1 + oy * 0.125f * r3) * (float)Hh - 0.5f;
        const float x0f = floorf(x), y0f = floorf(y);
        const float wx1 = x - x0f, wy1 = y - y0f;
        const float wx0 = 1.f - wx1, wy0 = 1.f - wy1;
        const int x0 = (int)x0f, y0 = (int)y0f;
        const float* vb = value + ((size_t)b * LENV + st) * DM + h * 32 + lane;

        float smp = 0.f, g;
        g = (x0 >= 0 && x0 < Ww && y0 >= 0 && y0 < Hh)
                ? vb[(size_t)(y0 * Ww + x0) * DM] : 0.f;
        smp += wx0 * wy0 * g;
        g = (x0 + 1 >= 0 && x0 + 1 < Ww && y0 >= 0 && y0 < Hh)
                ? vb[(size_t)(y0 * Ww + x0 + 1) * DM] : 0.f;
        smp += wx1 * wy0 * g;
        g = (x0 >= 0 && x0 < Ww && y0 + 1 >= 0 && y0 + 1 < Hh)
                ? vb[(size_t)((y0 + 1) * Ww + x0) * DM] : 0.f;
        smp += wx0 * wy1 * g;
        g = (x0 + 1 >= 0 && x0 + 1 < Ww && y0 + 1 >= 0 && y0 + 1 < Hh)
                ? vb[(size_t)((y0 + 1) * Ww + x0 + 1) * DM] : 0.f;
        smp += wx1 * wy1 * g;

        acc += wp * smp;
    }
    ca[row * DM + h * 32 + lane] = acc;
}

// ---------------------------------------------------------------------------
// out = rms_norm(clip?(x + res)) * w  (wave per row), with optional fused
// f16 side outputs: outh = f16(out), oph = f16(out + pos).
// ---------------------------------------------------------------------------
__global__ __launch_bounds__(256)
void resnorm_kernel(const float* __restrict__ x, const float* __restrict__ res,
                    const float* __restrict__ w, float* __restrict__ out,
                    _Float16* __restrict__ outh, const float* __restrict__ pos,
                    _Float16* __restrict__ oph, int doClip)
{
    const int row  = blockIdx.x * 8 + (threadIdx.x >> 5);
    const int lane = threadIdx.x & 31;
    const int c0   = lane * 8;

    const float* xp = x   + (size_t)row * DM + c0;
    const float* rp = res + (size_t)row * DM + c0;
    float4 a0 = *(const float4*)xp;
    float4 a1 = *(const float4*)(xp + 4);
    float4 b0 = *(const float4*)rp;
    float4 b1 = *(const float4*)(rp + 4);

    float v[8] = {a0.x + b0.x, a0.y + b0.y, a0.z + b0.z, a0.w + b0.w,
                  a1.x + b1.x, a1.y + b1.y, a1.z + b1.z, a1.w + b1.w};
    if (doClip) {
        #pragma unroll
        for (int i = 0; i < 8; i++) v[i] = fminf(fmaxf(v[i], -65504.f), 65504.f);
    }
    float ss = 0.f;
    #pragma unroll
    for (int i = 0; i < 8; i++) ss += v[i] * v[i];
    ss += __shfl_xor(ss, 16, 32);
    ss += __shfl_xor(ss, 8, 32);
    ss += __shfl_xor(ss, 4, 32);
    ss += __shfl_xor(ss, 2, 32);
    ss += __shfl_xor(ss, 1, 32);
    const float inv = rsqrtf(ss * (1.0f / 256.0f) + 1e-6f);

    float o[8];
    #pragma unroll
    for (int i = 0; i < 8; i++) o[i] = v[i] * inv * w[c0 + i];
    #pragma unroll
    for (int i = 0; i < 8; i++) out[(size_t)row * DM + c0 + i] = o[i];
    if (outh) {
        #pragma unroll
        for (int i = 0; i < 8; i++)
            outh[(size_t)row * DM + c0 + i] = (_Float16)o[i];
    }
    if (oph) {
        const float* pp = pos + (size_t)row * DM + c0;
        float4 p0 = *(const float4*)pp;
        float4 p1 = *(const float4*)(pp + 4);
        const float pa[8] = {p0.x, p0.y, p0.z, p0.w, p1.x, p1.y, p1.z, p1.w};
        #pragma unroll
        for (int i = 0; i < 8; i++)
            oph[(size_t)row * DM + c0 + i] = (_Float16)(o[i] + pa[i]);
    }
}

// ---------------------------------------------------------------------------
// xg = f16( silu(x12[:, :512]) * x12[:, 512:] )
// ---------------------------------------------------------------------------
__global__ __launch_bounds__(256)
void silu_gate_kernel(const float* __restrict__ x12, _Float16* __restrict__ xg,
                      int total)
{
    const int i = blockIdx.x * 256 + threadIdx.x;
    if (i < total) {
        const int row = i >> 9;
        const int c   = i & 511;
        const float x1 = x12[(size_t)row * 1024 + c];
        const float x2 = x12[(size_t)row * 1024 + 512 + c];
        xg[i] = (_Float16)((x1 / (1.f + __expf(-x1))) * x2);
    }
}

// ---------------------------------------------------------------------------
// Host-side orchestration
// ---------------------------------------------------------------------------
static void launch_gemm(const _Float16* Ah, const _Float16* Wh, const float* bias,
                        float* C, int N, int K, hipStream_t s)
{
    dim3 grid(N / 64, MROWS / 64);
    gemm_bias_wmma<<<grid, 256, 0, s>>>(Ah, Wh, bias, C, N, K, MROWS);
}

static void launch_pack(const float* a, const float* b, _Float16* out, int total,
                        hipStream_t s)
{
    pack_f16_kernel<<<(total / 4 + 255) / 256, 256, 0, s>>>(a, b, out, total);
}

extern "C" void kernel_launch(void* const* d_in, const int* in_sizes, int n_in,
                              void* d_out, int out_size, void* d_ws, size_t ws_size,
                              hipStream_t stream)
{
    (void)in_sizes; (void)n_in; (void)out_size; (void)ws_size;

    const float* target = (const float*)d_in[0];
    const float* refpts = (const float*)d_in[1];
    const float* value  = (const float*)d_in[2];
    /* d_in[3] spatial_shapes: compile-time constant here */
    const float* pos    = (const float*)d_in[4];
    const float* in_w   = (const float*)d_in[5];
    const float* in_b   = (const float*)d_in[6];
    const float* op_w   = (const float*)d_in[7];
    const float* op_b   = (const float*)d_in[8];
    const float* n1w    = (const float*)d_in[9];
    const float* n2w    = (const float*)d_in[10];
    const float* n3w    = (const float*)d_in[11];
    const float* so_w   = (const float*)d_in[12];
    const float* so_b   = (const float*)d_in[13];
    const float* aw_w   = (const float*)d_in[14];
    const float* aw_b   = (const float*)d_in[15];
    const float* w12    = (const float*)d_in[16];
    const float* b12    = (const float*)d_in[17];
    const float* w3     = (const float*)d_in[18];
    const float* b3     = (const float*)d_in[19];

    float* ws = (float*)d_ws;
    const size_t M = MROWS;

    // f32 region: q|k|v (reused as x12 [M,1024]), t1, off, aw, ca, t2
    float* qb   = ws;
    float* kb   = ws + M * 256;
    float* vb   = ws + 2 * M * 256;
    float* x12  = ws;                        // q/k/v + proj region reused
    float* proj = ws + 3 * M * 256;          // out-proj result
    float* t1   = ws + M * 1024;
    float* offb = t1 + M * 256;
    float* awb  = offb + M * 256;
    float* cab  = awb + M * 128;
    float* t2   = cab + M * 256;
    float* ffn  = cab;                       // ca dead after t2

    // f16 region
    _Float16* hb    = (_Float16*)(t2 + M * 256);
    _Float16* ahqk  = hb;                    // f16(target+pos)
    _Float16* aht   = ahqk + M * 256;        // f16(target)
    _Float16* sabh  = aht + M * 256;         // f16(self-attn out), by mha
    _Float16* qch   = sabh + M * 256;        // f16(t1+pos), by resnorm1
    _Float16* t2h   = qch + M * 256;         // f16(t2), by resnorm2
    _Float16* xgh   = t2h + M * 256;         // f16(silu gate), M*512
    _Float16* inw_h = xgh + M * 512;         // packed weights
    _Float16* opw_h = inw_h + 768 * 256;
    _Float16* sow_h = opw_h + 256 * 256;
    _Float16* aww_h = sow_h + 256 * 256;
    _Float16* w12_h = aww_h + 128 * 256;
    _Float16* w3_h  = w12_h + 1024 * 256;

    // 0) pack activations + weights to f16 (once; ~3MB of traffic)
    launch_pack(target, pos,     ahqk, MROWS * 256, stream);
    launch_pack(target, nullptr, aht,  MROWS * 256, stream);
    launch_pack(in_w, nullptr, inw_h, 768 * 256, stream);
    launch_pack(op_w, nullptr, opw_h, 256 * 256, stream);
    launch_pack(so_w, nullptr, sow_h, 256 * 256, stream);
    launch_pack(aw_w, nullptr, aww_h, 128 * 256, stream);
    launch_pack(w12,  nullptr, w12_h, 1024 * 256, stream);
    launch_pack(w3,   nullptr, w3_h,  256 * 512, stream);

    // 1) QKV projections (Q,K from target+pos; V from target)
    launch_gemm(ahqk, inw_h,             in_b,       qb, 256, 256, stream);
    launch_gemm(ahqk, inw_h + 256 * 256, in_b + 256, kb, 256, 256, stream);
    launch_gemm(aht,  inw_h + 512 * 256, in_b + 512, vb, 256, 256, stream);

    // 2) self-attention -> f16 sa
    mha_kernel<<<dim3(19, NH, BQ), 64, 0, stream>>>(qb, kb, vb, sabh);

    // 3) out projection + residual RMSNorm1 -> t1 (+ fused f16(t1+pos))
    launch_gemm(sabh, opw_h, op_b, proj, 256, 256, stream);
    resnorm_kernel<<<MROWS / 8, 256, 0, stream>>>(proj, target, n1w, t1,
                                                  nullptr, pos, qch, 0);

    // 4) sampling offsets + attention weights
    launch_gemm(qch, sow_h, so_b, offb, 256, 256, stream);
    launch_gemm(qch, aww_h, aw_b, awb,  128, 256, stream);

    // 5) deformable gather -> ca, then RMSNorm2 -> t2 (+ fused f16(t2))
    deform_attn<<<MROWS * NH / 8, 256, 0, stream>>>(offb, awb, refpts, value, cab);
    resnorm_kernel<<<MROWS / 8, 256, 0, stream>>>(cab, t1, n2w, t2,
                                                  t2h, nullptr, nullptr, 0);

    // 6) FFN: w12 -> silu gate (f16) -> w3, then clip+residual RMSNorm3 -> out
    launch_gemm(t2h, w12_h, b12, x12, 1024, 256, stream);
    silu_gate_kernel<<<(MROWS * 512 + 255) / 256, 256, 0, stream>>>(x12, xgh,
                                                                    MROWS * 512);
    launch_gemm(xgh, w3_h, b3, ffn, 256, 512, stream);
    resnorm_kernel<<<MROWS / 8, 256, 0, stream>>>(ffn, t2, n3w, (float*)d_out,
                                                  nullptr, nullptr, nullptr, 1);
}
